// MultiHeadAttention_15006615732561
// MI455X (gfx1250) — matrix operand order repaired and checked by reading, not compiled, not run on hardware
//
#include <hip/hip_runtime.h>
#include <hip/hip_bf16.h>

typedef __attribute__((ext_vector_type(16))) _Float16 v16h;
typedef __attribute__((ext_vector_type(8)))  _Float16 v8h;
typedef __attribute__((ext_vector_type(4)))  _Float16 v4h;
typedef __attribute__((ext_vector_type(8)))  float    v8f;

#define LDS_PAD   40   // f16 per LDS row (32 data + 8 pad) = 80B = 20 banks, 16B aligned
#define BLK_M    256
#define BLK_N     64
#define BLK_K     32

// ---------------------------------------------------------------------------
// Double-buffered LDS-staged WMMA GEMM:  C[z] = scale * (A[z] @ B[z]) + bias
//   A: [M,K] row-major (lda), f32
//   B: BT=0: [K,N] row-major (ldb);  BT=1: stored [N,K] row-major, used as B^T
//   C: [M,N] row-major (ldc), f32
// Block = 256 threads (8 waves) -> 256x64 output tile, K-step 32.
// Each wave: 32x64 strip = 2 A-frags x 4 B-frags = 8 WMMA per K-step.
// LDS (f16): Alds[m][k], Blds[n][k], row pad 40 -> every lane fragment is
// two contiguous 16B chunks (ds_load_b128), matching ISA 7.12.2 layouts.
// Stage k+1 while computing k (one barrier per iteration); prefetch k+2.
// Grids must tile M,N,K exactly (all launches below do).
// ---------------------------------------------------------------------------
template <bool BT>
__global__ __launch_bounds__(256)
void gemm_wmma_f16(const float* __restrict__ A, long lda, long sAb, long sAh,
                   const float* __restrict__ B, long ldb, long sBb, long sBh,
                   float*       __restrict__ C, long ldc, long sCb, long sCh,
                   const float* __restrict__ bias, float scale,
                   int M, int N, int K, int H)
{
    __shared__ _Float16 Alds[2][BLK_M * LDS_PAD];   // 2 x 20480 B
    __shared__ _Float16 Blds[2][BLK_N * LDS_PAD];   // 2 x  5120 B

    const int tid  = threadIdx.x;
    const int wave = tid >> 5;
    const int lane = tid & 31;
    const int half = lane >> 4;
    const int lm   = lane & 15;

    const int blkM = blockIdx.y * BLK_M;
    const int blkN = blockIdx.x * BLK_N;

    const int zb = blockIdx.z / H;
    const int zh = blockIdx.z % H;
    A += zb * sAb + zh * sAh;
    B += zb * sBb + zh * sBh;
    C += zb * sCb + zh * sCh;

    v8f acc[2][4] = {};

    // ---- stage one K-tile into LDS buffer `buf` (all loads batched first) --
    auto stage = [&](int k0, int buf) {
        float4 fa[8];
        #pragma unroll
        for (int i = 0; i < 8; ++i) {              // A: 2048 float4 slots
            const int slot = tid + i * 256;
            const int row  = slot >> 3;            // 0..255
            const int kq   = slot & 7;
            fa[i] = *(const float4*)(A + (long)(blkM + row) * lda + k0 + kq * 4);
        }
        float4 fb[2];
        #pragma unroll
        for (int i = 0; i < 2; ++i) {              // B: 512 float4 slots
            const int slot = tid + i * 256;
            if (BT) {                              // B stored [N,K]
                const int n  = slot >> 3;
                const int kq = slot & 7;
                fb[i] = *(const float4*)(B + (long)(blkN + n) * ldb + k0 + kq * 4);
            } else {                               // B stored [K,N]
                const int k  = slot >> 4;
                const int nq = slot & 15;
                fb[i] = *(const float4*)(B + (long)(k0 + k) * ldb + blkN + nq * 4);
            }
        }
        #pragma unroll
        for (int i = 0; i < 8; ++i) {
            const int slot = tid + i * 256;
            const int row  = slot >> 3;
            const int kq   = slot & 7;
            v4h h; h[0] = (_Float16)fa[i].x; h[1] = (_Float16)fa[i].y;
                   h[2] = (_Float16)fa[i].z; h[3] = (_Float16)fa[i].w;
            *(v4h*)(&Alds[buf][row * LDS_PAD + kq * 4]) = h;
        }
        #pragma unroll
        for (int i = 0; i < 2; ++i) {
            const int slot = tid + i * 256;
            if (BT) {                              // direct copy, contiguous K
                const int n  = slot >> 3;
                const int kq = slot & 7;
                v4h h; h[0] = (_Float16)fb[i].x; h[1] = (_Float16)fb[i].y;
                       h[2] = (_Float16)fb[i].z; h[3] = (_Float16)fb[i].w;
                *(v4h*)(&Blds[buf][n * LDS_PAD + kq * 4]) = h;
            } else {                               // transpose scatter to [n][k]
                const int k  = slot >> 4;
                const int nq = slot & 15;
                Blds[buf][(nq * 4 + 0) * LDS_PAD + k] = (_Float16)fb[i].x;
                Blds[buf][(nq * 4 + 1) * LDS_PAD + k] = (_Float16)fb[i].y;
                Blds[buf][(nq * 4 + 2) * LDS_PAD + k] = (_Float16)fb[i].z;
                Blds[buf][(nq * 4 + 3) * LDS_PAD + k] = (_Float16)fb[i].w;
            }
        }
    };

    // ---- compute one K-step from LDS buffer `buf`: 8 WMMA per wave --------
    auto compute = [&](int buf) {
        v16h a[2];
        #pragma unroll
        for (int m = 0; m < 2; ++m) {
            const _Float16* Ar = &Alds[buf][(wave * 32 + m * 16 + lm) * LDS_PAD];
            const v8h a0 = *(const v8h*)(Ar + half * 8);
            const v8h a1 = *(const v8h*)(Ar + 16 + half * 8);
            #pragma unroll
            for (int e = 0; e < 8; ++e) { a[m][e] = a0[e]; a[m][8 + e] = a1[e]; }
        }
        #pragma unroll
        for (int t = 0; t < 4; ++t) {
            const _Float16* Br = &Blds[buf][(t * 16 + lm) * LDS_PAD];
            const v8h b0 = *(const v8h*)(Br + half * 8);
            const v8h b1 = *(const v8h*)(Br + 16 + half * 8);
            v16h bfrag;
            #pragma unroll
            for (int e = 0; e < 8; ++e) { bfrag[e] = b0[e]; bfrag[8 + e] = b1[e]; }
            acc[0][t] = __builtin_amdgcn_wmma_f32_16x16x32_f16(
                            false, a[0], false, bfrag, (short)0, acc[0][t], false, false);
            acc[1][t] = __builtin_amdgcn_wmma_f32_16x16x32_f16(
                            false, a[1], false, bfrag, (short)0, acc[1][t], false, false);
        }
    };

    stage(0, 0);
    __syncthreads();

    int buf = 0;
    for (int k0 = 0; k0 < K; k0 += BLK_K) {
        if (k0 + BLK_K < K) {
            stage(k0 + BLK_K, buf ^ 1);            // overlap with compute(buf)
            if (k0 + 2 * BLK_K < K)                // prefetch two tiles ahead
                __builtin_prefetch(A + (long)(blkM + tid) * lda + k0 + 2 * BLK_K, 0, 1);
        }
        compute(buf);
        buf ^= 1;
        __syncthreads();
    }

    // ---- store C (16x16 f32 layout: VGPR r -> row r + 8*half, col = lm) ----
    #pragma unroll
    for (int t = 0; t < 4; ++t) {
        const int col = blkN + t * 16 + lm;
        const float bv = bias ? bias[col] : 0.0f;
        #pragma unroll
        for (int m = 0; m < 2; ++m) {
            #pragma unroll
            for (int r = 0; r < 8; ++r) {
                const int row = blkM + wave * 32 + m * 16 + r + 8 * half;
                C[(long)row * ldc + col] = acc[m][t][r] * scale + bv;
            }
        }
    }
}

// ---------------------------------------------------------------------------
// Row softmax, in place. One wave per 2048-wide row, 8 rows per block.
// float4 (b128) coalesced I/O; register-resident; wave32 shfl reductions.
// ---------------------------------------------------------------------------
__global__ __launch_bounds__(256)
void softmax_rows2048(float* __restrict__ P, long numRows)
{
    const int  wave = threadIdx.x >> 5;
    const int  lane = threadIdx.x & 31;
    const long row  = (long)blockIdx.x * 8 + wave;
    if (row >= numRows) return;
    float4* p4 = (float4*)(P + row * 2048);

    float4 v[16];
    float mx = -__builtin_inff();
    #pragma unroll
    for (int i = 0; i < 16; ++i) {
        v[i] = p4[lane + i * 32];
        mx = fmaxf(mx, fmaxf(fmaxf(v[i].x, v[i].y), fmaxf(v[i].z, v[i].w)));
    }
    #pragma unroll
    for (int off = 16; off > 0; off >>= 1)
        mx = fmaxf(mx, __shfl_xor(mx, off, 32));

    float sum = 0.0f;
    #pragma unroll
    for (int i = 0; i < 16; ++i) {
        v[i].x = __expf(v[i].x - mx); v[i].y = __expf(v[i].y - mx);
        v[i].z = __expf(v[i].z - mx); v[i].w = __expf(v[i].w - mx);
        sum += (v[i].x + v[i].y) + (v[i].z + v[i].w);
    }
    #pragma unroll
    for (int off = 16; off > 0; off >>= 1)
        sum += __shfl_xor(sum, off, 32);

    const float inv = 1.0f / sum;
    #pragma unroll
    for (int i = 0; i < 16; ++i) {
        v[i].x *= inv; v[i].y *= inv; v[i].z *= inv; v[i].w *= inv;
        p4[lane + i * 32] = v[i];
    }
}

extern "C" void kernel_launch(void* const* d_in, const int* in_sizes, int n_in,
                              void* d_out, int out_size, void* d_ws, size_t ws_size,
                              hipStream_t stream)
{
    const float* query = (const float*)d_in[0];
    const float* key_  = (const float*)d_in[1];
    const float* value = (const float*)d_in[2];
    const float* wq    = (const float*)d_in[3];
    const float* bq    = (const float*)d_in[4];
    const float* wk    = (const float*)d_in[5];
    const float* bk    = (const float*)d_in[6];
    const float* wv    = (const float*)d_in[7];
    const float* bvv   = (const float*)d_in[8];
    const float* wf    = (const float*)d_in[9];
    const float* bf    = (const float*)d_in[10];

    const int  Bb = 2, S = 2048, D = 1024, H = 16;
    const long MS = (long)Bb * S;            // 4096 rows total
    const long SD = (long)S * D;             // per-batch projected size
    const long SS = (long)S * S;             // per-head score matrix

    float* ws   = (float*)d_ws;              // needs 4 * 16 MB = 64 MB
    float* Qp   = ws;                        // [B,S,D]
    float* Kp   = Qp + MS * D;
    float* Vp   = Kp + MS * D;
    float* attn = Vp + MS * D;               // [B,S,D] (already head-merged)

    float* out   = (float*)d_out;            // [B,S,D]
    float* probs = out + MS * D;             // [B,H,S,S]

    dim3 blk(256);

    // --- 1) projections: [4096,1024] @ [1024,1024] + bias ---
    dim3 gProj(D / BLK_N, (unsigned)(MS / BLK_M), 1);
    gemm_wmma_f16<false><<<gProj, blk, 0, stream>>>(
        query, D, 0, 0,  wq, D, 0, 0,  Qp, D, 0, 0,  bq, 1.0f,
        (int)MS, D, D, 1);
    gemm_wmma_f16<false><<<gProj, blk, 0, stream>>>(
        key_,  D, 0, 0,  wk, D, 0, 0,  Kp, D, 0, 0,  bk, 1.0f,
        (int)MS, D, D, 1);
    gemm_wmma_f16<false><<<gProj, blk, 0, stream>>>(
        value, D, 0, 0,  wv, D, 0, 0,  Vp, D, 0, 0,  bvv, 1.0f,
        (int)MS, D, D, 1);

    // --- 2) scores = Vp_head @ Kp_head^T / 8  -> probs region (raw) ---
    dim3 gScore(S / BLK_N, S / BLK_M, Bb * H);
    gemm_wmma_f16<true><<<gScore, blk, 0, stream>>>(
        Vp, D, SD, 64,                 // A: V head [S,64]
        Kp, D, SD, 64,                 // B^T: K head stored [S,64]
        probs, S, (long)H * SS, SS,    // C: [S,S] per (b,h)
        nullptr, 0.125f,
        S, S, 64, H);

    // --- 3) softmax over rows of probs (in place) ---
    const long rows = (long)Bb * H * S;      // 65536
    softmax_rows2048<<<(unsigned)(rows / 8), blk, 0, stream>>>(probs, rows);

    // --- 4) attn = probs @ Qp_head  -> attn ws as [B,S,H*64] ---
    dim3 gAttn(64 / BLK_N, S / BLK_M, Bb * H);
    gemm_wmma_f16<false><<<gAttn, blk, 0, stream>>>(
        probs, S, (long)H * SS, SS,    // A: [S,S] per (b,h)
        Qp, D, SD, 64,                 // B: Q head [S,64], ldb = 1024
        attn, D, SD, 64,               // C: head slice of [B,S,1024]
        nullptr, 1.0f,
        S, 64, S, H);

    // --- 5) out = attn @ wf + bf ---
    gemm_wmma_f16<false><<<gProj, blk, 0, stream>>>(
        attn, D, 0, 0,  wf, D, 0, 0,  out, D, 0, 0,  bf, 1.0f,
        (int)MS, D, D, 1);
}
